// Attention_7696581394614
// MI455X (gfx1250) — matrix operand order repaired and checked by reading, not compiled, not run on hardware
//
#include <hip/hip_runtime.h>
#include <hip/hip_bf16.h>

// CDNA5 / gfx1250 attention block: QKV proj -> flash attention -> out proj.
// All GEMMs use v_wmma_f32_16x16x32_f16 (f16 operands, f32 accumulate).
// wave32; 256-thread blocks = 8 waves; all fragments load as ds_load_b128.
// LDS tiles are double-buffered: exactly one s_barrier pair per K-step.

typedef _Float16 half_t;
typedef __attribute__((ext_vector_type(16))) _Float16 v16h;
typedef __attribute__((ext_vector_type(8)))  float    v8f;

union U16h { v16h v; uint4 u[2]; };

#define DIM      768
#define HEADS    12
#define HEAD_DIM 64
#define BATCH    4
#define SEQ      2048
#define ROWS     (BATCH * SEQ)      /* 8192 */
#define QKV_N    (3 * DIM)          /* 2304 */
#define ATT_SCALE 0.125f            /* 64^-0.5 */

// ---------------------------------------------------------------- conversion
__global__ void k_cvt(const float* __restrict__ s, half_t* __restrict__ d, int n) {
    int i = blockIdx.x * blockDim.x + threadIdx.x;
    if (i < n) d[i] = (half_t)s[i];
}

// ---------------------------------------------------------------- QKV GEMM
// C[row,c] = sum_k xh[row,k] * W[c,k] + bias[c]  (W kept native [N][K]).
// Block tile 64(M) x 128(N) x 32(K); 8 waves in 2x4 grid, 32x32 per wave.
__global__ __launch_bounds__(256) void k_qkv(
    const half_t* __restrict__ xh, const half_t* __restrict__ wh,
    const float* __restrict__ bias,
    half_t* __restrict__ Qh, half_t* __restrict__ Kh, half_t* __restrict__ Vt) {
    __shared__ __align__(16) half_t at[2][64][40];    // A tile, M-major
    __shared__ __align__(16) half_t bt[2][128][40];   // B tile, N-major

    int bid  = blockIdx.x;
    int m0   = (bid / (QKV_N / 128)) * 64;
    int n0   = (bid % (QKV_N / 128)) * 128;
    int tid  = threadIdx.x;
    int lane = tid & 31, wave = tid >> 5;
    int wm   = wave >> 2, wn = wave & 3;
    int lo   = lane & 15, hi = lane >> 4;

    int sr = tid >> 2, ss = (tid & 3) * 8;            // staging row / segment

    uint4 aR, bR0, bR1;
    auto fetch = [&](int k0) {
        aR  = *(const uint4*)&xh[(size_t)(m0 + sr) * DIM + k0 + ss];
        bR0 = *(const uint4*)&wh[(size_t)(n0 + sr) * DIM + k0 + ss];
        bR1 = *(const uint4*)&wh[(size_t)(n0 + 64 + sr) * DIM + k0 + ss];
    };
    fetch(0);

    v8f acc[2][2] = {};
    int buf = 0;
    for (int k0 = 0; k0 < DIM; k0 += 32, buf ^= 1) {
        *(uint4*)&at[buf][sr][ss]      = aR;
        *(uint4*)&bt[buf][sr][ss]      = bR0;
        *(uint4*)&bt[buf][64 + sr][ss] = bR1;
        __syncthreads();                              // publish tile
        if (k0 + 32 < DIM) fetch(k0 + 32);            // overlap with WMMAs

        U16h a[2], b[2];
#pragma unroll
        for (int i = 0; i < 2; ++i) {
            const half_t* ap = &at[buf][wm * 32 + i * 16 + lo][0];
            a[i].u[0] = *(const uint4*)(ap + hi * 8);
            a[i].u[1] = *(const uint4*)(ap + 16 + hi * 8);
            const half_t* bp = &bt[buf][wn * 32 + i * 16 + lo][0];
            b[i].u[0] = *(const uint4*)(bp + hi * 8);
            b[i].u[1] = *(const uint4*)(bp + 16 + hi * 8);
        }
#pragma unroll
        for (int i = 0; i < 2; ++i)
#pragma unroll
            for (int j = 0; j < 2; ++j)
                acc[i][j] = __builtin_amdgcn_wmma_f32_16x16x32_f16(
                    false, a[i].v, false, b[j].v, (short)0, acc[i][j], false, false);
    }

    // epilogue: bias + scatter per reference qkv.reshape(B,P,3,hd,H) layout
#pragma unroll
    for (int i = 0; i < 2; ++i)
#pragma unroll
        for (int j = 0; j < 2; ++j)
#pragma unroll
            for (int r = 0; r < 8; ++r) {
                int row = m0 + wm * 32 + i * 16 + hi * 8 + r;
                int c   = n0 + wn * 32 + j * 16 + lo;
                int bb  = row >> 11, p = row & 2047;
                int t   = c / DIM, rem = c % DIM;
                int dd  = rem / HEADS, hh = rem % HEADS;
                half_t hv = (half_t)(acc[i][j][r] + bias[c]);
                size_t hb = (size_t)bb * HEADS + hh;
                if (t == 0)      Qh[(hb * SEQ + p) * HEAD_DIM + dd] = hv;
                else if (t == 1) Kh[(hb * SEQ + p) * HEAD_DIM + dd] = hv;
                else             Vt[(hb * HEAD_DIM + dd) * SEQ + p] = hv; // V^T
            }
}

// ---------------------------------------------------------------- attention
// One block per (b, h, 128-query tile); 8 waves x 16 query rows.
// K kept natural [p][d] (score B-frag is N-major); V transposed [d][p].
__global__ __launch_bounds__(256) void k_attn(
    const half_t* __restrict__ Qh, const half_t* __restrict__ Kh,
    const half_t* __restrict__ Vt, half_t* __restrict__ Oh) {
    __shared__ __align__(16) half_t kt[2][32][72];    // keys x d
    __shared__ __align__(16) half_t vt[2][64][40];    // d x keys
    __shared__ __align__(16) half_t pt[8][16][40];    // per-wave exp(S) relayout

    int bid  = blockIdx.x;
    int qt   = bid & 15;
    int bh   = bid >> 4;                 // 0..47
    int b    = bh / HEADS, h = bh % HEADS;
    int q0   = qt * 128;
    int tid  = threadIdx.x;
    int lane = tid & 31, wave = tid >> 5;
    int lo   = lane & 15, hi = lane >> 4;

    // Q fragments, vector-loaded, live in registers for the whole key loop
    const half_t* qp =
        &Qh[((size_t)bh * SEQ + (q0 + wave * 16 + lo)) * HEAD_DIM];
    U16h qa0, qa1;
    qa0.u[0] = *(const uint4*)(qp + hi * 8);
    qa0.u[1] = *(const uint4*)(qp + 16 + hi * 8);
    qa1.u[0] = *(const uint4*)(qp + 32 + hi * 8);
    qa1.u[1] = *(const uint4*)(qp + 48 + hi * 8);

    int kr = tid >> 3, ks = (tid & 7) * 8;   // K staging: 32 rows x 64
    int vr = tid >> 2, vs = (tid & 3) * 8;   // V staging: 64 rows x 32
    uint4 kR, vR;
    auto fetch = [&](int j0) {
        kR = *(const uint4*)&Kh[((size_t)bh * SEQ + j0 + kr) * HEAD_DIM + ks];
        vR = *(const uint4*)&Vt[((size_t)bh * HEAD_DIM + vr) * SEQ + j0 + vs];
    };
    fetch(0);

    v8f o0 = {}, o1 = {}, o2 = {}, o3 = {};
    float rm[8], rl[8];
#pragma unroll
    for (int r = 0; r < 8; ++r) { rm[r] = -1e30f; rl[r] = 0.f; }

    int buf = 0;
    for (int j0 = 0; j0 < SEQ; j0 += 32, buf ^= 1) {
        *(uint4*)&kt[buf][kr][ks] = kR;
        *(uint4*)&vt[buf][vr][vs] = vR;
        __syncthreads();                     // publish K/V tile
        if (j0 + 32 < SEQ) fetch(j0 + 32);

        // scores: S0 = keys +0..15, S1 = keys +16..31 (K-dim = hd = 64)
        v8f s0 = {}, s1 = {};
#pragma unroll
        for (int dh = 0; dh < 2; ++dh) {
            U16h kb;
            const half_t* kp = &kt[buf][lo][dh * 32 + hi * 8];
            kb.u[0] = *(const uint4*)kp;
            kb.u[1] = *(const uint4*)(kp + 16);
            s0 = __builtin_amdgcn_wmma_f32_16x16x32_f16(
                false, dh ? qa1.v : qa0.v, false, kb.v, (short)0, s0, false, false);
            kp = &kt[buf][16 + lo][dh * 32 + hi * 8];
            kb.u[0] = *(const uint4*)kp;
            kb.u[1] = *(const uint4*)(kp + 16);
            s1 = __builtin_amdgcn_wmma_f32_16x16x32_f16(
                false, dh ? qa1.v : qa0.v, false, kb.v, (short)0, s1, false, false);
        }

        // online softmax (row stats reduced across the 16-lane halves)
        float tm[8];
#pragma unroll
        for (int r = 0; r < 8; ++r)
            tm[r] = fmaxf(s0[r] * ATT_SCALE, s1[r] * ATT_SCALE);
#pragma unroll
        for (int off = 1; off < 16; off <<= 1) {
#pragma unroll
            for (int r = 0; r < 8; ++r)
                tm[r] = fmaxf(tm[r], __shfl_xor(tm[r], off, 32));
        }
        float ps[8];
#pragma unroll
        for (int r = 0; r < 8; ++r) {
            float nm   = fmaxf(rm[r], tm[r]);
            float corr = __expf(rm[r] - nm);
            rm[r] = nm;
            rl[r] *= corr;
            o0[r] *= corr; o1[r] *= corr; o2[r] *= corr; o3[r] *= corr;
            float p0 = __expf(s0[r] * ATT_SCALE - nm);
            float p1 = __expf(s1[r] * ATT_SCALE - nm);
            pt[wave][hi * 8 + r][lo]      = (half_t)p0;
            pt[wave][hi * 8 + r][16 + lo] = (half_t)p1;
            ps[r] = p0 + p1;
        }
#pragma unroll
        for (int off = 1; off < 16; off <<= 1) {
#pragma unroll
            for (int r = 0; r < 8; ++r) ps[r] += __shfl_xor(ps[r], off, 32);
        }
#pragma unroll
        for (int r = 0; r < 8; ++r) rl[r] += ps[r];

        // pt is wave-private; same-wave LDS ops are in-order (DScnt), so only
        // a compiler scheduling fence is needed -- no hardware barrier.
        __builtin_amdgcn_wave_barrier();

        U16h pa;
        const half_t* pp = &pt[wave][lo][0];
        pa.u[0] = *(const uint4*)(pp + hi * 8);
        pa.u[1] = *(const uint4*)(pp + 16 + hi * 8);

        // O += P @ V   (N = hd = 64 -> 4 accumulator tiles)
#pragma unroll
        for (int dt = 0; dt < 4; ++dt) {
            U16h vb;
            const half_t* vp = &vt[buf][dt * 16 + lo][hi * 8];
            vb.u[0] = *(const uint4*)vp;
            vb.u[1] = *(const uint4*)(vp + 16);
            v8f& o = dt == 0 ? o0 : dt == 1 ? o1 : dt == 2 ? o2 : o3;
            o = __builtin_amdgcn_wmma_f32_16x16x32_f16(
                false, pa.v, false, vb.v, (short)0, o, false, false);
        }
    }

    // normalize + write merged [B, P, h*64 + d] layout (f16 for proj GEMM)
#pragma unroll
    for (int r = 0; r < 8; ++r) {
        float inv = 1.0f / rl[r];
        int q = q0 + wave * 16 + hi * 8 + r;
        size_t ob = ((size_t)b * SEQ + q) * DIM + h * HEAD_DIM;
        Oh[ob +  0 + lo] = (half_t)(o0[r] * inv);
        Oh[ob + 16 + lo] = (half_t)(o1[r] * inv);
        Oh[ob + 32 + lo] = (half_t)(o2[r] * inv);
        Oh[ob + 48 + lo] = (half_t)(o3[r] * inv);
    }
}

// ---------------------------------------------------------------- out proj
__global__ __launch_bounds__(256) void k_proj(
    const half_t* __restrict__ Oh, const half_t* __restrict__ wh,
    const float* __restrict__ bias, float* __restrict__ out) {
    __shared__ __align__(16) half_t at[2][64][40];
    __shared__ __align__(16) half_t bt[2][128][40];

    int bid  = blockIdx.x;
    int m0   = (bid / (DIM / 128)) * 64;
    int n0   = (bid % (DIM / 128)) * 128;
    int tid  = threadIdx.x;
    int lane = tid & 31, wave = tid >> 5;
    int wm   = wave >> 2, wn = wave & 3;
    int lo   = lane & 15, hi = lane >> 4;

    int sr = tid >> 2, ss = (tid & 3) * 8;

    uint4 aR, bR0, bR1;
    auto fetch = [&](int k0) {
        aR  = *(const uint4*)&Oh[(size_t)(m0 + sr) * DIM + k0 + ss];
        bR0 = *(const uint4*)&wh[(size_t)(n0 + sr) * DIM + k0 + ss];
        bR1 = *(const uint4*)&wh[(size_t)(n0 + 64 + sr) * DIM + k0 + ss];
    };
    fetch(0);

    v8f acc[2][2] = {};
    int buf = 0;
    for (int k0 = 0; k0 < DIM; k0 += 32, buf ^= 1) {
        *(uint4*)&at[buf][sr][ss]      = aR;
        *(uint4*)&bt[buf][sr][ss]      = bR0;
        *(uint4*)&bt[buf][64 + sr][ss] = bR1;
        __syncthreads();
        if (k0 + 32 < DIM) fetch(k0 + 32);

        U16h a[2], b[2];
#pragma unroll
        for (int i = 0; i < 2; ++i) {
            const half_t* ap = &at[buf][wm * 32 + i * 16 + lo][0];
            a[i].u[0] = *(const uint4*)(ap + hi * 8);
            a[i].u[1] = *(const uint4*)(ap + 16 + hi * 8);
            const half_t* bp = &bt[buf][wn * 32 + i * 16 + lo][0];
            b[i].u[0] = *(const uint4*)(bp + hi * 8);
            b[i].u[1] = *(const uint4*)(bp + 16 + hi * 8);
        }
#pragma unroll
        for (int i = 0; i < 2; ++i)
#pragma unroll
            for (int j = 0; j < 2; ++j)
                acc[i][j] = __builtin_amdgcn_wmma_f32_16x16x32_f16(
                    false, a[i].v, false, b[j].v, (short)0, acc[i][j], false, false);
    }
#pragma unroll
    for (int i = 0; i < 2; ++i)
#pragma unroll
        for (int j = 0; j < 2; ++j)
#pragma unroll
            for (int r = 0; r < 8; ++r) {
                int row = m0 + wm * 32 + i * 16 + hi * 8 + r;
                int c   = n0 + wn * 32 + j * 16 + lo;
                out[(size_t)row * DIM + c] = acc[i][j][r] + bias[c];
            }
}

// ---------------------------------------------------------------- launch
extern "C" void kernel_launch(void* const* d_in, const int* in_sizes, int n_in,
                              void* d_out, int out_size, void* d_ws, size_t ws_size,
                              hipStream_t stream) {
    (void)in_sizes; (void)n_in; (void)out_size; (void)ws_size;
    const float* x      = (const float*)d_in[0];
    const float* w_qkv  = (const float*)d_in[1];
    const float* b_qkv  = (const float*)d_in[2];
    const float* w_proj = (const float*)d_in[3];
    const float* b_proj = (const float*)d_in[4];
    float* out = (float*)d_out;

    // f16 workspace carve-out (~65 MB total)
    half_t* p      = (half_t*)d_ws;
    half_t* xh     = p; p += (size_t)ROWS * DIM;
    half_t* wqkvh  = p; p += (size_t)QKV_N * DIM;   // native [N][K]
    half_t* wprojh = p; p += (size_t)DIM * DIM;     // native [N][K]
    half_t* Qh     = p; p += (size_t)BATCH * HEADS * SEQ * HEAD_DIM;
    half_t* Kh     = p; p += (size_t)BATCH * HEADS * SEQ * HEAD_DIM;
    half_t* Vt     = p; p += (size_t)BATCH * HEADS * SEQ * HEAD_DIM; // [d][p]
    half_t* Oh     = p;

    int n;
    n = ROWS * DIM;
    k_cvt<<<(n + 255) / 256, 256, 0, stream>>>(x, xh, n);
    n = QKV_N * DIM;
    k_cvt<<<(n + 255) / 256, 256, 0, stream>>>(w_qkv, wqkvh, n);
    n = DIM * DIM;
    k_cvt<<<(n + 255) / 256, 256, 0, stream>>>(w_proj, wprojh, n);

    k_qkv<<<(ROWS / 64) * (QKV_N / 128), 256, 0, stream>>>(xh, wqkvh, b_qkv,
                                                           Qh, Kh, Vt);
    k_attn<<<BATCH * HEADS * (SEQ / 128), 256, 0, stream>>>(Qh, Kh, Vt, Oh);
    k_proj<<<(ROWS / 64) * (DIM / 128), 256, 0, stream>>>(Oh, wprojh, b_proj, out);
}
